// MPNNModel_42417097015744
// MI455X (gfx1250) — compile-verified, matrix operand order checked by reading, not run on hardware
//
#include <hip/hip_runtime.h>
#include <hip/hip_bf16.h>
#include <stdint.h>

typedef __attribute__((ext_vector_type(16))) _Float16 v16h;
typedef __attribute__((ext_vector_type(8)))  float    v8f;
typedef __attribute__((ext_vector_type(8)))  unsigned int v8u;

#define N_NODES  50000
#define N_GRAPHS 512
#define DF       128   // all feature dims are 128

// ---------------- generic fill ----------------
__global__ void fill_kernel(float* __restrict__ p, float v, int n) {
    for (int i = blockIdx.x * blockDim.x + threadIdx.x; i < n; i += gridDim.x * blockDim.x)
        p[i] = v;
}

// ---------------- degree via atomics (dst side), deg pre-initialized to 1.0 (self loop) ----------------
__global__ void deg_kernel(const int* __restrict__ dst, float* __restrict__ deg, int E) {
    int e = blockIdx.x * blockDim.x + threadIdx.x;
    if (e < E) atomicAdd(&deg[dst[e]], 1.0f);
}

__global__ void rsqrt_kernel(float* __restrict__ d, int n) {
    int i = blockIdx.x * blockDim.x + threadIdx.x;
    if (i < n) d[i] = rsqrtf(d[i]);   // deg >= 1 always (self loops)
}

// ---------------- f32 -> f16 convert ----------------
__global__ void f32_to_f16_kernel(const float* __restrict__ in, _Float16* __restrict__ out, int n) {
    int i = blockIdx.x * blockDim.x + threadIdx.x;
    if (i < n) out[i] = (_Float16)in[i];
}

// ---------------- pack W[128x128] f32 into per-lane WMMA B fragments (f16 pairs) ----------------
// idx = ntile*1024 + kstep*256 + lane*8 + v   (8 ntiles * 4 ksteps * 32 lanes * 8 dwords = 8192)
__global__ void pack_w_kernel(const float* __restrict__ W, uint32_t* __restrict__ Wp) {
    int idx = blockIdx.x * blockDim.x + threadIdx.x;
    if (idx >= 8192) return;
    int v     =  idx        & 7;
    int lane  = (idx >> 3)  & 31;
    int kstep = (idx >> 8)  & 3;
    int ntile =  idx >> 10;
    int khalf = lane >> 4;
    int col   = ntile * 16 + (lane & 15);
    int kb    = ((v < 4) ? 2 * v : 2 * v + 8) + 8 * khalf + 32 * kstep;
    _Float16 lo = (_Float16)W[kb * DF + col];
    _Float16 hi = (_Float16)W[(kb + 1) * DF + col];
    uint32_t ulo = (uint32_t)__builtin_bit_cast(unsigned short, lo);
    uint32_t uhi = (uint32_t)__builtin_bit_cast(unsigned short, hi);
    Wp[idx] = ulo | (uhi << 16);
}

// ---------------- WMMA GEMM: C[M,128] = A[M,128](f16) @ W(packed f16), f32 accum ----------------
// block = (32,8): one wave per (mtile, ntile); grid.x = M/16 (M = 50000 -> exactly 3125 tiles)
__global__ __launch_bounds__(256) void gemm_wmma_kernel(const _Float16* __restrict__ A,
                                                        const uint32_t* __restrict__ Wp,
                                                        float* __restrict__ C) {
    int lane  = threadIdx.x;          // 0..31
    int ntile = threadIdx.y;          // 0..7
    int m0    = blockIdx.x * 16;
    int khalf = lane >> 4;
    int row   = m0 + (lane & 15);
    const uint32_t* Arow = (const uint32_t*)A + (size_t)row * (DF / 2);

    v8f acc = {};
#pragma unroll
    for (int ks = 0; ks < 4; ++ks) {
        const uint32_t* wp = Wp + (((size_t)ntile * 4 + ks) * 32 + lane) * 8;
        v8u au, bu;
#pragma unroll
        for (int v = 0; v < 8; ++v) {
            int kb = ((v < 4) ? 2 * v : 2 * v + 8) + 8 * khalf + 32 * ks;  // even
            au[v] = Arow[kb >> 1];
            bu[v] = wp[v];
        }
        v16h a = __builtin_bit_cast(v16h, au);
        v16h b = __builtin_bit_cast(v16h, bu);
        acc = __builtin_amdgcn_wmma_f32_16x16x32_f16(false, a, false, b,
                                                     (short)0, acc, false, false);
    }

    int n  = ntile * 16 + (lane & 15);
    int mb = m0 + 8 * khalf;
#pragma unroll
    for (int v = 0; v < 8; ++v)
        C[(size_t)(mb + v) * DF + n] = acc[v];
}

// ---------------- self-loop contribution: full (non-atomic) init of h_out ----------------
// one wave per node, float4 per lane (32*4 = 128 channels)
__global__ void selfloop_init_kernel(const float* __restrict__ h, const float* __restrict__ dinv,
                                     float* __restrict__ out, int n) {
    int t = blockIdx.x * blockDim.x + threadIdx.x;
    int node = t >> 5;
    if (node >= n) return;
    int lane = t & 31;
    float w = dinv[node] * dinv[node];
    float4 v = ((const float4*)(h + (size_t)node * DF))[lane];
    float4 r; r.x = w * v.x; r.y = w * v.y; r.z = w * v.z; r.w = w * v.w;
    ((float4*)(out + (size_t)node * DF))[lane] = r;
}

// ---------------- edge scatter: out[dst] += dinv[src]*dinv[dst] * h[src] ----------------
// one wave per edge; atomics land in L2 (h_out is 25.6 MB << 192 MB L2)
__global__ void edge_scatter_kernel(const int* __restrict__ src, const int* __restrict__ dst,
                                    const float* __restrict__ dinv, const float* __restrict__ h,
                                    float* __restrict__ out, int E) {
    int t = blockIdx.x * blockDim.x + threadIdx.x;
    int e = t >> 5;
    if (e >= E) return;
    int lane = t & 31;
    int s = src[e], d = dst[e];
    float w = dinv[s] * dinv[d];
    float4 v = ((const float4*)(h + (size_t)s * DF))[lane];
    float* od = out + (size_t)d * DF + lane * 4;
    atomicAdd(od + 0, w * v.x);
    atomicAdd(od + 1, w * v.y);
    atomicAdd(od + 2, w * v.z);
    atomicAdd(od + 3, w * v.w);
}

// ---------------- bias + ReLU + convert to f16 (next-layer input) ----------------
__global__ void bias_relu_f16_kernel(const float* __restrict__ h, const float* __restrict__ b,
                                     _Float16* __restrict__ out, int n) {
    int i = blockIdx.x * blockDim.x + threadIdx.x;
    if (i >= n) return;
    float t = h[i] + b[i & (DF - 1)];
    out[i] = (_Float16)fmaxf(t, 0.0f);
}

// ---------------- bias in place (last layer, no ReLU) ----------------
__global__ void bias_kernel(float* __restrict__ h, const float* __restrict__ b, int n) {
    int i = blockIdx.x * blockDim.x + threadIdx.x;
    if (i < n) h[i] += b[i & (DF - 1)];
}

// ---------------- global add pool: out[batch[i]] += h[i] ----------------
__global__ void pool_kernel(const float* __restrict__ h, const int* __restrict__ batch,
                            float* __restrict__ out, int n) {
    int t = blockIdx.x * blockDim.x + threadIdx.x;
    int node = t >> 5;
    if (node >= n) return;
    int lane = t & 31;
    int g = batch[node];
    float4 v = ((const float4*)(h + (size_t)node * DF))[lane];
    float* og = out + (size_t)g * DF + lane * 4;
    atomicAdd(og + 0, v.x);
    atomicAdd(og + 1, v.y);
    atomicAdd(og + 2, v.z);
    atomicAdd(og + 3, v.w);
}

extern "C" void kernel_launch(void* const* d_in, const int* in_sizes, int n_in,
                              void* d_out, int out_size, void* d_ws, size_t ws_size,
                              hipStream_t stream) {
    const float* x     = (const float*)d_in[0];
    const int*   ei    = (const int*)  d_in[1];   // [2, E]: row0 = src, row1 = dst
    const int*   batch = (const int*)  d_in[2];
    const float* W1 = (const float*)d_in[3]; const float* b1 = (const float*)d_in[4];
    const float* W2 = (const float*)d_in[5]; const float* b2 = (const float*)d_in[6];
    const float* W3 = (const float*)d_in[7]; const float* b3 = (const float*)d_in[8];

    const int N = in_sizes[0] / DF;        // 50000 (exactly 3125 * 16)
    const int E = in_sizes[1] / 2;         // 800000
    const int*   src = ei;
    const int*   dst = ei + E;

    // -------- workspace layout (256B aligned) : ~64.3 MB --------
    char* ws = (char*)d_ws;
    float*    dinv   = (float*)ws;     ws += (((size_t)N * 4) + 255) & ~(size_t)255;
    uint32_t* Wp     = (uint32_t*)ws;  ws += 8192 * 4;
    _Float16* feat16 = (_Float16*)ws;  ws += (size_t)N * DF * 2;        // 12.8 MB (256-mult.)
    float*    h_lin  = (float*)ws;     ws += (size_t)N * DF * 4;        // 25.6 MB
    float*    h_out  = (float*)ws;

    const int B = 256;
    // degrees (init 1.0 for self loop) -> dinv
    fill_kernel<<<1024, B, 0, stream>>>(dinv, 1.0f, N);
    deg_kernel<<<(E + B - 1) / B, B, 0, stream>>>(dst, dinv, E);
    rsqrt_kernel<<<(N + B - 1) / B, B, 0, stream>>>(dinv, N);

    // first-layer input
    f32_to_f16_kernel<<<(N * DF + B - 1) / B, B, 0, stream>>>(x, feat16, N * DF);

    const float* Ws[3] = {W1, W2, W3};
    const float* bs[3] = {b1, b2, b3};
    const int mtiles    = N / 16;                 // 3125
    const int edgeBlks  = (E * 32 + B - 1) / B;   // wave per edge
    const int nodeBlks  = (N * 32 + B - 1) / B;   // wave per node

    for (int l = 0; l < 3; ++l) {
        pack_w_kernel<<<8192 / B, B, 0, stream>>>(Ws[l], Wp);
        gemm_wmma_kernel<<<mtiles, dim3(32, 8), 0, stream>>>(feat16, Wp, h_lin);
        selfloop_init_kernel<<<nodeBlks, B, 0, stream>>>(h_lin, dinv, h_out, N);
        edge_scatter_kernel<<<edgeBlks, B, 0, stream>>>(src, dst, dinv, h_lin, h_out, E);
        if (l < 2)
            bias_relu_f16_kernel<<<(N * DF + B - 1) / B, B, 0, stream>>>(h_out, bs[l], feat16, N * DF);
        else
            bias_kernel<<<(N * DF + B - 1) / B, B, 0, stream>>>(h_out, bs[l], N * DF);
    }

    // global add pool
    fill_kernel<<<64, B, 0, stream>>>((float*)d_out, 0.0f, N_GRAPHS * DF);
    pool_kernel<<<nodeBlks, B, 0, stream>>>(h_out, batch, (float*)d_out, N);
}
